// ProtoNet_76149770158544
// MI455X (gfx1250) — matrix operand order repaired and checked by reading, not compiled, not run on hardware
//
#include <hip/hip_runtime.h>
#include <hip/hip_bf16.h>

typedef __attribute__((ext_vector_type(16))) _Float16 v16h;
typedef __attribute__((ext_vector_type(8)))  _Float16 v8h;
typedef __attribute__((ext_vector_type(8)))  float    v8f;

#define NS   32768
#define NQ   65536
#define NWAY 64
#define D0   768
#define D1   512
#define D2   512
#define D3   128
#define D4   32
#define LN_EPS 1e-5f
#define MBLK 32          // rows per workgroup
#define THREADS 256      // 8 wave32

// ---------------------------------------------------------------------------
// WMMA helpers
// ---------------------------------------------------------------------------
static __device__ inline v8f wmma16(v16h a, v16h b, v8f c) {
  // D = A(16x32 f16) x B(32x16 f16) + C(16x16 f32)
  return __builtin_amdgcn_wmma_f32_16x16x32_f16(
      /*neg_a=*/false, a, /*neg_b=*/false, b,
      /*c_mod=*/(short)0, c, /*reuse_a=*/false, /*reuse_b=*/false);
}

// A-fragment, contiguous k order, same as load_b (16x32, M x K):
//  lanes 0-15 : row M=lane,     K=kc*32+0..15
//  lanes 16-31: row M=lane-16,  K=kc*32+16..31
static __device__ inline v16h load_a(const _Float16* base, int stride,
                                     int m0, int kc, int lane) {
  int row = m0 + (lane & 15);
  int kb  = kc * 32 + ((lane & 16) ? 16 : 0);
  const _Float16* p = base + row * stride + kb;
  union { v16h v; v8h h[2]; } u;
  u.h[0] = *(const v8h*)(p);        // 16B aligned LDS load
  u.h[1] = *(const v8h*)(p + 8);    // 16B aligned LDS load
  return u.v;
}

// B-fragment (32x16, K x N) from row-major f16 W[N][K]:
//  lanes 0-15 : col N=n0+lane,    K = kc*32+0..15  (contiguous in W row)
//  lanes 16-31: col N=n0+lane-16, K = kc*32+16..31
static __device__ inline v16h load_b(const _Float16* W, int K,
                                     int n0, int kc, int lane) {
  const _Float16* p = W + (size_t)(n0 + (lane & 15)) * K
                        + kc * 32 + ((lane & 16) ? 16 : 0);
  return *(const v16h*)p;           // 32B aligned global load
}

// ---------------------------------------------------------------------------
// Fused GEMM (out = relu(LN(in @ W^T + b)) ) for one 32-row block.
// N split over 8 waves, NTW 16-wide N-tiles per wave; K-accumulate in regs.
// ---------------------------------------------------------------------------
template<int K, int N, int NTW>
static __device__ void gemm_ln_relu(const _Float16* in, int inStride,
                                    const _Float16* Wg, const float* bias,
                                    const float* gamma, const float* beta,
                                    _Float16* out, int outStride,
                                    float* sSum, float* sSumSq, int tid) {
  const int wave = tid >> 5;
  const int lane = tid & 31;

  v8f acc[2][NTW];
#pragma unroll
  for (int mt = 0; mt < 2; ++mt)
#pragma unroll
    for (int nt = 0; nt < NTW; ++nt)
      acc[mt][nt] = (v8f){};

  for (int kc = 0; kc < K / 32; ++kc) {
    v16h a0 = load_a(in, inStride, 0, kc, lane);
    v16h a1 = load_a(in, inStride, 16, kc, lane);
#pragma unroll
    for (int nt = 0; nt < NTW; ++nt) {
      int n0 = (wave * NTW + nt) * 16;
      v16h b = load_b(Wg, K, n0, kc, lane);
      acc[0][nt] = wmma16(a0, b, acc[0][nt]);
      acc[1][nt] = wmma16(a1, b, acc[1][nt]);
    }
  }

  __syncthreads();                      // all reads of `in` done -> may alias out
  if (tid < MBLK) { sSum[tid] = 0.f; sSumSq[tid] = 0.f; }
  __syncthreads();

  // bias add + per-row partial stats (C-layout: VGPR r, lanes0-15 -> M=r,
  // lanes16-31 -> M=8+r; col = ntile*16 + lane%16)
  float colb[NTW];
#pragma unroll
  for (int nt = 0; nt < NTW; ++nt)
    colb[nt] = bias[(wave * NTW + nt) * 16 + (lane & 15)];

#pragma unroll
  for (int mt = 0; mt < 2; ++mt) {
#pragma unroll
    for (int r = 0; r < 8; ++r) {
      float s = 0.f, s2 = 0.f;
#pragma unroll
      for (int nt = 0; nt < NTW; ++nt) {
        float v = acc[mt][nt][r] + colb[nt];
        acc[mt][nt][r] = v;
        s += v; s2 += v * v;
      }
      // butterfly over each 16-lane half (masks <16 stay within half)
#pragma unroll
      for (int m = 1; m < 16; m <<= 1) {
        s  += __shfl_xor(s,  m, 32);
        s2 += __shfl_xor(s2, m, 32);
      }
      int row = (mt << 4) + r + ((lane & 16) ? 8 : 0);
      if ((lane & 15) == 0) {
        atomicAdd(&sSum[row],   s);     // ds_add_f32
        atomicAdd(&sSumSq[row], s2);
      }
    }
  }
  __syncthreads();

  if (tid < MBLK) {
    float mu  = sSum[tid] * (1.0f / N);
    float var = sSumSq[tid] * (1.0f / N) - mu * mu;
    sSum[tid]   = mu;
    sSumSq[tid] = rsqrtf(var + LN_EPS);
  }
  __syncthreads();

#pragma unroll
  for (int mt = 0; mt < 2; ++mt)
#pragma unroll
    for (int nt = 0; nt < NTW; ++nt) {
      int col = (wave * NTW + nt) * 16 + (lane & 15);
      float g = gamma[col], be = beta[col];
#pragma unroll
      for (int r = 0; r < 8; ++r) {
        int row = (mt << 4) + r + ((lane & 16) ? 8 : 0);
        float v = (acc[mt][nt][r] - sSum[row]) * sSumSq[row] * g + be;
        v = fmaxf(v, 0.f);
        out[row * outStride + col] = (_Float16)v;
      }
    }
  __syncthreads();
}

// ---------------------------------------------------------------------------
// Fused backbone: 32 rows/block through all 4 layers; z -> global f32
// ---------------------------------------------------------------------------
__global__ __launch_bounds__(THREADS)
void k_backbone(const float* __restrict__ sup, const float* __restrict__ qry,
                const _Float16* __restrict__ W1h, const _Float16* __restrict__ W2h,
                const _Float16* __restrict__ W3h, const _Float16* __restrict__ W4h,
                const float* __restrict__ b1, const float* __restrict__ g1, const float* __restrict__ be1,
                const float* __restrict__ b2, const float* __restrict__ g2, const float* __restrict__ be2,
                const float* __restrict__ b3, const float* __restrict__ g3, const float* __restrict__ be3,
                const float* __restrict__ b4, float* __restrict__ zAll) {
  __shared__ _Float16 smem[MBLK * (D0 + 8)];   // reused: X(776) -> h(520) -> h(136)
  __shared__ float sSum[MBLK], sSumSq[MBLK];

  const int tid  = threadIdx.x;
  const int lane = tid & 31;
  const int wave = tid >> 5;
  const int row0 = blockIdx.x * MBLK;

  // load + convert 32x768 f32 -> f16 LDS tile (row stride 776)
  const float* src = (row0 < NS) ? (sup + (size_t)row0 * D0)
                                 : (qry + (size_t)(row0 - NS) * D0);
  for (int i = tid; i < MBLK * D0; i += THREADS) {
    int r = i / D0, c = i - r * D0;
    smem[r * (D0 + 8) + c] = (_Float16)src[(size_t)r * D0 + c];
  }
  __syncthreads();

  gemm_ln_relu<D0, D1, 4>(smem, D0 + 8, W1h, b1, g1, be1, smem, D1 + 8, sSum, sSumSq, tid);
  gemm_ln_relu<D1, D2, 4>(smem, D1 + 8, W2h, b2, g2, be2, smem, D2 + 8, sSum, sSumSq, tid);
  gemm_ln_relu<D2, D3, 1>(smem, D2 + 8, W3h, b3, g3, be3, smem, D3 + 8, sSum, sSumSq, tid);

  // layer 4: 32x128 @ (32x128)^T -> 32x32, no LN; waves 0..3, 1 tile each
  if (wave < 4) {
    int mt = wave >> 1, nt = wave & 1;
    v8f acc = (v8f){};
    for (int kc = 0; kc < D3 / 32; ++kc) {
      v16h a = load_a(smem, D3 + 8, mt * 16, kc, lane);
      v16h b = load_b(W4h, D3, nt * 16, kc, lane);
      acc = wmma16(a, b, acc);
    }
    float cb = b4[nt * 16 + (lane & 15)];
#pragma unroll
    for (int r = 0; r < 8; ++r) {
      int row = mt * 16 + r + ((lane & 16) ? 8 : 0);
      int col = nt * 16 + (lane & 15);
      zAll[(size_t)(row0 + row) * D4 + col] = acc[r] + cb;
    }
  }
}

// ---------------------------------------------------------------------------
// Small kernels
// ---------------------------------------------------------------------------
__global__ void k_f32_to_f16(const float* __restrict__ s, _Float16* __restrict__ d, int n) {
  int i = blockIdx.x * 256 + threadIdx.x;
  if (i < n) d[i] = (_Float16)s[i];
}

__global__ void k_zero(float* __restrict__ p, int n) {
  for (int i = blockIdx.x * 256 + threadIdx.x; i < n; i += gridDim.x * 256) p[i] = 0.f;
}

__global__ void k_proto_accum(const float* __restrict__ zs,
                              const long long* __restrict__ labels,
                              float* __restrict__ proto, float* __restrict__ cnt) {
  int idx = blockIdx.x * 256 + threadIdx.x;       // NS*32 threads
  int r = idx >> 5, c = idx & 31;
  int lab = (int)labels[r];
  atomicAdd(&proto[lab * D4 + c], zs[idx]);
  if (c == 0) atomicAdd(&cnt[lab], 1.0f);
}

__global__ void k_proto_div(float* __restrict__ proto, const float* __restrict__ cnt) {
  int i = blockIdx.x * 256 + threadIdx.x;         // NWAY*32
  if (i < NWAY * D4) proto[i] /= fmaxf(cnt[i >> 5], 1.0f);
}

__global__ __launch_bounds__(256)
void k_dist(const float* __restrict__ zq, const float* __restrict__ proto,
            float* __restrict__ out) {
  __shared__ float sp[NWAY * D4];
  for (int i = threadIdx.x; i < NWAY * D4; i += 256) sp[i] = proto[i];
  __syncthreads();
  int gid = blockIdx.x * 256 + threadIdx.x;       // NQ*NWAY
  int q = gid >> 6, p = gid & 63;
  const float* zr = zq + (size_t)q * D4;
  float s = 0.f;
#pragma unroll
  for (int c = 0; c < D4; ++c) {
    float d = zr[c] - sp[p * D4 + c];
    s += d * d;
  }
  out[gid] = -sqrtf(fmaxf(s, 0.f));
}

// ---------------------------------------------------------------------------
// launch
// ---------------------------------------------------------------------------
extern "C" void kernel_launch(void* const* d_in, const int* in_sizes, int n_in,
                              void* d_out, int out_size, void* d_ws, size_t ws_size,
                              hipStream_t stream) {
  const float*     sup  = (const float*)d_in[0];
  const long long* lab  = (const long long*)d_in[1];
  const float*     qry  = (const float*)d_in[2];
  const float *W1 = (const float*)d_in[3],  *b1 = (const float*)d_in[4];
  const float *g1 = (const float*)d_in[5],  *be1= (const float*)d_in[6];
  const float *W2 = (const float*)d_in[7],  *b2 = (const float*)d_in[8];
  const float *g2 = (const float*)d_in[9],  *be2= (const float*)d_in[10];
  const float *W3 = (const float*)d_in[11], *b3 = (const float*)d_in[12];
  const float *g3 = (const float*)d_in[13], *be3= (const float*)d_in[14];
  const float *W4 = (const float*)d_in[15], *b4 = (const float*)d_in[16];
  float* out = (float*)d_out;

  // workspace layout
  char* ws = (char*)d_ws;
  _Float16* W1h = (_Float16*)ws;                    // 512*768
  _Float16* W2h = W1h + D1 * D0;                    // 512*512
  _Float16* W3h = W2h + D2 * D1;                    // 128*512
  _Float16* W4h = W3h + D3 * D2;                    // 32*128
  float* zAll  = (float*)(ws + 2u * 1024u * 1024u);           // (NS+NQ)*32 f32 = 12 MB
  float* proto = (float*)(ws + 2u * 1024u * 1024u + 13u * 1024u * 1024u);
  float* cnt   = proto + NWAY * D4;

  // 1) weights -> f16 (L2-resident, 1.45 MB total)
  k_f32_to_f16<<<(D1*D0 + 255) / 256, 256, 0, stream>>>(W1, W1h, D1*D0);
  k_f32_to_f16<<<(D2*D1 + 255) / 256, 256, 0, stream>>>(W2, W2h, D2*D1);
  k_f32_to_f16<<<(D3*D2 + 255) / 256, 256, 0, stream>>>(W3, W3h, D3*D2);
  k_f32_to_f16<<<(D4*D3 + 255) / 256, 256, 0, stream>>>(W4, W4h, D4*D3);

  // 2) fused backbone over all 98304 rows
  k_backbone<<<(NS + NQ) / MBLK, THREADS, 0, stream>>>(
      sup, qry, W1h, W2h, W3h, W4h,
      b1, g1, be1, b2, g2, be2, b3, g3, be3, b4, zAll);

  // 3) prototypes: zero, segment-sum, divide
  k_zero<<<9, 256, 0, stream>>>(proto, NWAY * D4 + NWAY);
  k_proto_accum<<<(NS * D4) / 256, 256, 0, stream>>>(zAll, lab, proto, cnt);
  k_proto_div<<<(NWAY * D4 + 255) / 256, 256, 0, stream>>>(proto, cnt);

  // 4) -cdist(query z, proto)
  k_dist<<<(NQ * NWAY) / 256, 256, 0, stream>>>(zAll + (size_t)NS * D4, proto, out);
}
